// EquivSetConv_49658411876804
// MI455X (gfx1250) — matrix operand order repaired and checked by reading, not compiled, not run on hardware
//
#include <hip/hip_runtime.h>

typedef __attribute__((ext_vector_type(16))) _Float16 v16h;
typedef __attribute__((ext_vector_type(8)))  float    v8f;

#define NN    50000
#define NEE   10000
#define NNZK  600000
#define DF    128   // DIN == DOUT == 128

// ---------------------------------------------------------------------------
// Fragment index maps for V_WMMA_F32_16X16X32_F16 (CDNA5 ISA §7.12.2).
// A (16x32 f16): lane m<16 holds K=0..7 (vgpr0-3) and K=16..23 (vgpr4-7);
//                lanes 16-31 hold K=8..15 and K=24..31.
// B (32x16 f16): same K order as A (lane n<16 holds K=0..7, 16..23;
//                lanes 16-31 hold K=8..15, 24..31).
// v16h element h -> vgpr j=h/2, half=h&1.
// ---------------------------------------------------------------------------
__device__ __forceinline__ int a_kidx(int lane, int h) {
    int j = h >> 1, half = h & 1;
    int k = (j < 4 ? 2 * j : 8 + 2 * j) + half;
    return k + ((lane >= 16) ? 8 : 0);
}
__device__ __forceinline__ int b_kidx(int lane, int h) {
    return (h < 8 ? h : 8 + h) + ((lane >= 16) ? 8 : 0);
}

// ---------------------------------------------------------------------------
// Pre-swizzle a 128x128 (row stride ldw) f32 weight matrix into B-fragment
// order: P[((ntile*4 + kc)*32 + lane)*16 + h] = (f16) W[n*ldw + k].
// ---------------------------------------------------------------------------
__global__ void pack_w_kernel(const float* __restrict__ W, int ldw,
                              _Float16* __restrict__ P) {
    int t = blockIdx.x * blockDim.x + threadIdx.x;
    if (t >= 8 * 4 * 32 * 16) return;
    int h    = t & 15;
    int lane = (t >> 4) & 31;
    int kc   = (t >> 9) & 3;
    int nt   = t >> 11;
    int n = nt * 16 + (lane & 15);
    int k = kc * 32 + b_kidx(lane, h);
    P[t] = (_Float16)W[(size_t)n * ldw + k];
}

// ---------------------------------------------------------------------------
// WMMA GEMM: O[m,n] = sum_k A'[m,k] * Wt[n,k] (+bias), K = 128.
//   FINMIX:  A' = 0.5*(A + Srow[m]*(A3 + bcol[c])) + 0.5*A2   (final stage)
//   else:    A' = A
//   DUAL: second n-output sharing the staged A tile.
// 256 threads = 8 wave32; wave w owns n-tile w; block b owns m-tile b.
// M must be a multiple of 16 (50000 = 16*3125, 10000 = 16*625).
// ---------------------------------------------------------------------------
template <bool DUAL, bool FINMIX>
__global__ void __launch_bounds__(256)
gemm128_kernel(const float* __restrict__ A, const float* __restrict__ A2,
               const float* __restrict__ A3, const float* __restrict__ Srow,
               const float* __restrict__ bcol,
               const _Float16* __restrict__ P0, const float* __restrict__ b0,
               float* __restrict__ O0,
               const _Float16* __restrict__ P1, const float* __restrict__ b1,
               float* __restrict__ O1) {
    __shared__ _Float16 As[16][DF + 2];   // +2 halves: conflict-free gather
    const int tid = threadIdx.x;
    const int m0  = blockIdx.x * 16;

    // Stage 16x128 A tile (fp32 -> fp16), coalesced.
    for (int i = tid; i < 16 * DF; i += 256) {
        int r = i >> 7, c = i & 127;
        size_t idx = (size_t)(m0 + r) * DF + c;
        float v = A[idx];
        if (FINMIX)
            v = 0.5f * (v + Srow[m0 + r] * (A3[idx] + bcol[c])) + 0.5f * A2[idx];
        As[r][c] = (_Float16)v;
    }
    __syncthreads();

    const int wave = tid >> 5;        // n-tile index 0..7
    const int lane = tid & 31;
    const int m    = lane & 15;

    const v16h* B0 = (const v16h*)P0 + (size_t)wave * 4 * 32 + lane;
    const v16h* B1 = DUAL ? ((const v16h*)P1 + (size_t)wave * 4 * 32 + lane) : nullptr;

    v8f acc0 = {};
    v8f acc1 = {};
#pragma unroll
    for (int kc = 0; kc < 4; ++kc) {
        v16h a;
#pragma unroll
        for (int h = 0; h < 16; ++h)
            a[h] = As[m][kc * 32 + a_kidx(lane, h)];

        acc0 = __builtin_amdgcn_wmma_f32_16x16x32_f16(
            false, a, false, B0[kc * 32], (short)0, acc0, false, false);
        if (DUAL)
            acc1 = __builtin_amdgcn_wmma_f32_16x16x32_f16(
                false, a, false, B1[kc * 32], (short)0, acc1, false, false);
    }

    // C/D layout: vgpr r -> row m0 + r + (lane>=16 ? 8 : 0), col = wave*16+(lane&15)
    const int n    = wave * 16 + (lane & 15);
    const int rowb = m0 + ((lane >= 16) ? 8 : 0);
    const float bv0 = b0 ? b0[n] : 0.0f;
#pragma unroll
    for (int r = 0; r < 8; ++r)
        O0[(size_t)(rowb + r) * DF + n] = acc0[r] + bv0;
    if (DUAL) {
        const float bv1 = b1 ? b1[n] : 0.0f;
#pragma unroll
        for (int r = 0; r < 8; ++r)
            O1[(size_t)(rowb + r) * DF + n] = acc1[r] + bv1;
    }
}

// ---------------------------------------------------------------------------
// Xe[edges[k]] += alpha[k] * Xw1[vertex[k]]   (one wave32 per nonzero,
// float4 per lane = 128 features). Xe (5.1 MB) and Xw1 (25.6 MB) are
// L2-resident, so both the gather and the atomics stay on-chip.
// ---------------------------------------------------------------------------
__global__ void __launch_bounds__(256)
scatter_edge_kernel(const float* __restrict__ Xw1, const int* __restrict__ vertex,
                    const int* __restrict__ edges, const float* __restrict__ alpha,
                    float* __restrict__ Xe) {
    int k = blockIdx.x * 8 + (threadIdx.x >> 5);
    if (k >= NNZK) return;
    int l = threadIdx.x & 31;
    int v = vertex[k], e = edges[k];
    float a = alpha[k];
    float4 x = ((const float4*)(Xw1 + (size_t)v * DF))[l];
    float* dst = Xe + (size_t)e * DF + l * 4;
    atomicAdd(dst + 0, a * x.x);
    atomicAdd(dst + 1, a * x.y);
    atomicAdd(dst + 2, a * x.z);
    atomicAdd(dst + 3, a * x.w);
}

// ---------------------------------------------------------------------------
// Factored vertex scatter:
//   Xv[v] += alpha[k] * Y2[edges[k]]      (vector part)
//   S[v]  += alpha[k]                     (scalar part; Y1/b2 folded in later)
// ---------------------------------------------------------------------------
__global__ void __launch_bounds__(256)
scatter_vertex_kernel(const float* __restrict__ Y2, const int* __restrict__ vertex,
                      const int* __restrict__ edges, const float* __restrict__ alpha,
                      float* __restrict__ Xv, float* __restrict__ S) {
    int k = blockIdx.x * 8 + (threadIdx.x >> 5);
    if (k >= NNZK) return;
    int l = threadIdx.x & 31;
    int v = vertex[k], e = edges[k];
    float a = alpha[k];
    float4 y2 = ((const float4*)(Y2 + (size_t)e * DF))[l];
    float* dst = Xv + (size_t)v * DF + l * 4;
    atomicAdd(dst + 0, a * y2.x);
    atomicAdd(dst + 1, a * y2.y);
    atomicAdd(dst + 2, a * y2.z);
    atomicAdd(dst + 3, a * y2.w);
    if (l == 0) atomicAdd(S + v, a);
}

__global__ void zero_kernel(float* __restrict__ p, int n) {
    int i = blockIdx.x * blockDim.x + threadIdx.x;
    if (i < n) p[i] = 0.0f;
}

// ---------------------------------------------------------------------------
extern "C" void kernel_launch(void* const* d_in, const int* in_sizes, int n_in,
                              void* d_out, int out_size, void* d_ws, size_t ws_size,
                              hipStream_t stream) {
    const float* X     = (const float*)d_in[0];
    const int*   vert  = (const int*)d_in[1];
    const int*   edges = (const int*)d_in[2];
    const float* X0    = (const float*)d_in[3];
    const float* alpha = (const float*)d_in[4];
    const float* W1_w  = (const float*)d_in[5];
    const float* W1_b  = (const float*)d_in[6];
    const float* W2_w  = (const float*)d_in[7];   // [128, 256]
    const float* W2_b  = (const float*)d_in[8];
    const float* W_w   = (const float*)d_in[9];
    const float* W_b   = (const float*)d_in[10];
    float* out = (float*)d_out;

    // Workspace layout (floats): Xw1 | Y1 | Y2 | Xv | Xe | S | packed f16 weights
    float* ws  = (float*)d_ws;
    float* Xw1 = ws;
    float* Y1  = Xw1 + (size_t)NN * DF;
    float* Y2  = Y1  + (size_t)NN * DF;
    float* Xv  = Y2  + (size_t)NEE * DF;
    float* Xe  = Xv  + (size_t)NN * DF;
    float* S   = Xe  + (size_t)NEE * DF;
    _Float16* P     = (_Float16*)(S + NN);
    _Float16* P_W1  = P;
    _Float16* P_W2a = P + 1 * 16384;
    _Float16* P_W2b = P + 2 * 16384;
    _Float16* P_W   = P + 3 * 16384;

    // 1) zero the atomic accumulators (Xv | Xe | S are contiguous)
    {
        int n = (NN + NEE) * DF + NN;
        zero_kernel<<<(n + 255) / 256, 256, 0, stream>>>(Xv, n);
    }

    // 2) pre-swizzle weights into WMMA B-fragment order (f16)
    pack_w_kernel<<<64, 256, 0, stream>>>(W1_w,       DF,  P_W1);   // W1
    pack_w_kernel<<<64, 256, 0, stream>>>(W2_w,       256, P_W2a);  // W2[:, :128]
    pack_w_kernel<<<64, 256, 0, stream>>>(W2_w + 128, 256, P_W2b);  // W2[:, 128:]
    pack_w_kernel<<<64, 256, 0, stream>>>(W_w,        DF,  P_W);    // W

    // 3) Xw1 = X@W1^T + b1   and   Y1 = X@W2a^T   (one pass over X)
    gemm128_kernel<true, false><<<NN / 16, 256, 0, stream>>>(
        X, nullptr, nullptr, nullptr, nullptr,
        P_W1, W1_b, Xw1, P_W2a, nullptr, Y1);

    // 4) Xe = segment_sum(alpha * Xw1[vertex], edges)
    scatter_edge_kernel<<<(NNZK + 7) / 8, 256, 0, stream>>>(Xw1, vert, edges, alpha, Xe);

    // 5) Y2 = Xe @ W2b^T
    gemm128_kernel<false, false><<<NEE / 16, 256, 0, stream>>>(
        Xe, nullptr, nullptr, nullptr, nullptr,
        P_W2b, nullptr, Y2, nullptr, nullptr, nullptr);

    // 6) Xv += alpha * Y2[edges] scattered by vertex;  S[v] += alpha
    scatter_vertex_kernel<<<(NNZK + 7) / 8, 256, 0, stream>>>(
        Y2, vert, edges, alpha, Xv, S);

    // 7) out = (0.5*(Xv + S*(Y1 + b2)) + 0.5*X0) @ W^T + W_b
    gemm128_kernel<false, true><<<NN / 16, 256, 0, stream>>>(
        Xv, X0, Y1, S, W2_b,
        P_W, W_b, out, nullptr, nullptr, nullptr);
}
